// WeightedGATv2ConvGraphGymLayer_59596966199883
// MI455X (gfx1250) — matrix operand order repaired and checked by reading, not compiled, not run on hardware
//
#include <hip/hip_runtime.h>
#include <math.h>

// WeightedGATv2Conv for MI455X (gfx1250, wave32).
// GEMMs via v_wmma_f32_16x16x32_f16 (f16 in, f32 accumulate); edge phase with
// wave-per-edge float4 loads, shfl head-reduction, f32 global atomics for
// segment max / sum / scatter-add. All scratch lives in d_ws.

#define DD   256   // input dim == H*C
#define HH   8
#define AS_STRIDE 264  // 256 halves + 8 pad halves -> conflict-free b128 DS reads

typedef _Float16 v16h __attribute__((ext_vector_type(16)));
typedef float    v8f  __attribute__((ext_vector_type(8)));
typedef _Float16 h8   __attribute__((ext_vector_type(8)));

// ---------------- prep / init ----------------

__global__ void k_conv_x(const float* __restrict__ x, _Float16* __restrict__ xh, int total) {
  int t = blockIdx.x * blockDim.x + threadIdx.x;
  if (t < total) xh[t] = (_Float16)x[t];
}

// W is [K=256][N=256] row-major f32 -> wT is [N=256][K=256] f16 (so B-fragment
// reads are contiguous along K).
__global__ void k_conv_w(const float* __restrict__ wl, const float* __restrict__ wr,
                         _Float16* __restrict__ wlT, _Float16* __restrict__ wrT) {
  int t = blockIdx.x * blockDim.x + threadIdx.x;   // 0..65535
  int k = t >> 8, n = t & 255;
  wlT[n * 256 + k] = (_Float16)wl[t];
  wrT[n * 256 + k] = (_Float16)wr[t];
}

__global__ void k_init_out(float* __restrict__ out, const float* __restrict__ bias, int total) {
  int t = blockIdx.x * blockDim.x + threadIdx.x;
  if (t < total) out[t] = bias[t & 255];
}

__global__ void k_init_stats(float* __restrict__ amax, float* __restrict__ denom, int total) {
  int t = blockIdx.x * blockDim.x + threadIdx.x;
  if (t < total) { amax[t] = -INFINITY; denom[t] = 0.f; }
}

// ---------------- WMMA GEMM: x_l = x@W_l + b_l ; x_r = x@W_r + b_r ----------------
// Block = 256 threads = 8 waves, one 16-row M tile per block.
// Waves 0..3 -> W_l cols [64w, 64w+64); waves 4..7 -> W_r likewise.

__global__ __launch_bounds__(256) void k_gemm(
    const _Float16* __restrict__ xh,
    const _Float16* __restrict__ wlT, const _Float16* __restrict__ wrT,
    const float* __restrict__ bl, const float* __restrict__ br,
    float* __restrict__ xl, float* __restrict__ xr, int N)
{
  __shared__ __align__(16) _Float16 As[16 * AS_STRIDE];
  const int tid  = threadIdx.x;
  const int wave = tid >> 5;
  const int lane = tid & 31;
  const int row0 = blockIdx.x << 4;

  // Stage 16x256 f16 x-tile into LDS (512 x 16B, 2 per thread), padded rows.
  #pragma unroll
  for (int it = 0; it < 2; ++it) {
    int idx = tid + it * 256;          // 0..511
    int r = idx >> 5, s = idx & 31;    // 16 rows x 32 segments of 8 halves
    int gr = row0 + r; if (gr >= N) gr = N - 1;   // clamp (N%16==0 in practice)
    *(uint4*)(&As[r * AS_STRIDE + s * 8]) =
        *(const uint4*)(xh + (size_t)gr * 256 + s * 8);
  }
  __syncthreads();

  const bool isR = wave >= 4;
  const _Float16* wT = isR ? wrT : wlT;
  const float* bias  = isR ? br  : bl;
  float* y           = isR ? xr  : xl;
  const int col0 = (wave & 3) << 6;

  const int mlane = lane & 15;
  const int aoff  = (lane < 16) ? 0 : 8;    // A: K halves {aoff..aoff+7, aoff+16..aoff+23}
  const int koffB = (lane < 16) ? 0 : 8;    // B: K halves {koffB..koffB+7, koffB+16..koffB+23}

  v8f acc[4];
  #pragma unroll
  for (int t = 0; t < 4; ++t) acc[t] = (v8f){0.f,0.f,0.f,0.f,0.f,0.f,0.f,0.f};

  #pragma unroll
  for (int kb = 0; kb < 256; kb += 32) {
    v16h a;
    {
      h8 lo = *(const h8*)(&As[mlane * AS_STRIDE + kb + aoff]);
      h8 hi = *(const h8*)(&As[mlane * AS_STRIDE + kb + aoff + 16]);
      #pragma unroll
      for (int i = 0; i < 8; ++i) { a[i] = lo[i]; a[8 + i] = hi[i]; }
    }
    #pragma unroll
    for (int t = 0; t < 4; ++t) {
      const _Float16* bp = wT + (size_t)(col0 + t * 16 + mlane) * 256 + kb + koffB;
      h8 lo = *(const h8*)bp;
      h8 hi = *(const h8*)(bp + 16);
      v16h b;
      #pragma unroll
      for (int i = 0; i < 8; ++i) { b[i] = lo[i]; b[8 + i] = hi[i]; }
      acc[t] = __builtin_amdgcn_wmma_f32_16x16x32_f16(
          false, a, false, b, (short)0, acc[t], false, false);
    }
  }

  // D layout: VGPR v -> M = v (+8 for lanes>=16), N = lane%16.
  const int mb = row0 + ((lane < 16) ? 0 : 8);
  #pragma unroll
  for (int t = 0; t < 4; ++t) {
    int col = col0 + t * 16 + mlane;
    float bv = bias[col];
    #pragma unroll
    for (int v = 0; v < 8; ++v) {
      int r = mb + v;
      if (r < N) y[(size_t)r * 256 + col] = acc[t][v] + bv;
    }
  }
}

// ---------------- edge phase ----------------

__device__ inline void atomicMaxF(float* addr, float val) {
  if (val >= 0.f) atomicMax((int*)addr, __float_as_int(val));
  else            atomicMin((unsigned int*)addr, __float_as_uint(val));
}

// One wave32 per edge: lane covers channels [8*lane, 8*lane+8); head = lane/4.
__global__ __launch_bounds__(256) void k_edge_alpha(
    const float* __restrict__ xl, const float* __restrict__ xr,
    const int* __restrict__ ei, const float* __restrict__ ea,
    const float* __restrict__ att, float* __restrict__ alpha,
    float* __restrict__ amax, int E)
{
  int w = blockIdx.x * 8 + (threadIdx.x >> 5);
  if (w >= E) return;
  int lane = threadIdx.x & 31;
  int src = ei[w], dst = ei[E + w];
  int ch0 = lane << 3;

  const float4* pa = (const float4*)(xl + (size_t)src * 256 + ch0);
  const float4* pb = (const float4*)(xr + (size_t)dst * 256 + ch0);
  const float4* pt = (const float4*)(att + ch0);   // att flat [H*C] == channel index

  float s = 0.f;
  #pragma unroll
  for (int q = 0; q < 2; ++q) {
    float4 a = pa[q], b = pb[q], t = pt[q];
    float g;
    g = a.x + b.x; g = g > 0.f ? g : 0.2f * g; s += g * t.x;
    g = a.y + b.y; g = g > 0.f ? g : 0.2f * g; s += g * t.y;
    g = a.z + b.z; g = g > 0.f ? g : 0.2f * g; s += g * t.z;
    g = a.w + b.w; g = g > 0.f ? g : 0.2f * g; s += g * t.w;
  }
  // reduce over the 4 lanes of each head (lane bits [1:0])
  s += __shfl_xor(s, 1, 32);
  s += __shfl_xor(s, 2, 32);

  if ((lane & 3) == 0) {
    int h = lane >> 2;
    float al = s + logf(ea[w]);
    alpha[(size_t)w * HH + h] = al;
    atomicMaxF(&amax[(size_t)dst * HH + h], al);
  }
}

// One thread per (edge, head): ex = exp(alpha - amax[dst]); denom += ex.
__global__ void k_edge_exp(const int* __restrict__ ei, float* __restrict__ alpha,
                           const float* __restrict__ amax, float* __restrict__ denom, int E)
{
  int t = blockIdx.x * blockDim.x + threadIdx.x;
  if (t >= E * HH) return;
  int e = t >> 3, h = t & 7;
  int dst = ei[E + e];
  float ex = expf(alpha[t] - amax[(size_t)dst * HH + h]);
  alpha[t] = ex;                       // overwrite in place (re-written each call)
  atomicAdd(&denom[(size_t)dst * HH + h], ex);
}

// One wave32 per edge: out[dst] += x_l[src] * ex/denom  (8 f32 atomics per lane).
__global__ __launch_bounds__(256) void k_edge_agg(
    const float* __restrict__ xl, const int* __restrict__ ei,
    const float* __restrict__ exw, const float* __restrict__ denom,
    float* __restrict__ out, int E)
{
  int w = blockIdx.x * 8 + (threadIdx.x >> 5);
  if (w >= E) return;
  int lane = threadIdx.x & 31;
  int src = ei[w], dst = ei[E + w];
  int h = lane >> 2;
  float coeff = exw[(size_t)w * HH + h] /
                (denom[(size_t)dst * HH + h] + 1e-16f);
  int ch0 = lane << 3;
  const float4* pj = (const float4*)(xl + (size_t)src * 256 + ch0);
  float4 j0 = pj[0], j1 = pj[1];
  float* po = out + (size_t)dst * 256 + ch0;
  atomicAdd(po + 0, j0.x * coeff);
  atomicAdd(po + 1, j0.y * coeff);
  atomicAdd(po + 2, j0.z * coeff);
  atomicAdd(po + 3, j0.w * coeff);
  atomicAdd(po + 4, j1.x * coeff);
  atomicAdd(po + 5, j1.y * coeff);
  atomicAdd(po + 6, j1.z * coeff);
  atomicAdd(po + 7, j1.w * coeff);
}

// ---------------- launch ----------------

extern "C" void kernel_launch(void* const* d_in, const int* in_sizes, int n_in,
                              void* d_out, int out_size, void* d_ws, size_t ws_size,
                              hipStream_t stream) {
  (void)n_in; (void)out_size; (void)ws_size;
  const int N = in_sizes[0] / DD;
  const int E = in_sizes[2];

  const float* x    = (const float*)d_in[0];
  const int*   ei   = (const int*)  d_in[1];
  const float* ea   = (const float*)d_in[2];
  const float* Wl   = (const float*)d_in[3];
  const float* bl   = (const float*)d_in[4];
  const float* Wr   = (const float*)d_in[5];
  const float* br   = (const float*)d_in[6];
  const float* att  = (const float*)d_in[7];
  const float* bias = (const float*)d_in[8];
  float* out = (float*)d_out;

  // workspace carve-out (256B aligned segments)
  char* ws = (char*)d_ws;
  size_t off = 0;
  auto carve = [&](size_t bytes) -> char* {
    char* p = ws + off; off += (bytes + 255) & ~(size_t)255; return p;
  };
  _Float16* xh   = (_Float16*)carve((size_t)N * DD * sizeof(_Float16));
  _Float16* wlT  = (_Float16*)carve((size_t)DD * DD * sizeof(_Float16));
  _Float16* wrT  = (_Float16*)carve((size_t)DD * DD * sizeof(_Float16));
  float* xl      = (float*)carve((size_t)N * DD * sizeof(float));
  float* xr      = (float*)carve((size_t)N * DD * sizeof(float));
  float* alpha   = (float*)carve((size_t)E * HH * sizeof(float));
  float* amax    = (float*)carve((size_t)N * HH * sizeof(float));
  float* denom   = (float*)carve((size_t)N * HH * sizeof(float));

  const int blk = 256;

  // init + precision conversion
  k_init_out  <<<(N * DD + blk - 1) / blk, blk, 0, stream>>>(out, bias, N * DD);
  k_init_stats<<<(N * HH + blk - 1) / blk, blk, 0, stream>>>(amax, denom, N * HH);
  k_conv_x    <<<(N * DD + blk - 1) / blk, blk, 0, stream>>>(x, xh, N * DD);
  k_conv_w    <<<(DD * DD) / blk, blk, 0, stream>>>(Wl, Wr, wlT, wrT);

  // projections via WMMA
  k_gemm<<<(N + 15) / 16, blk, 0, stream>>>(xh, wlT, wrT, bl, br, xl, xr, N);

  // edge phase
  k_edge_alpha<<<(E + 7) / 8, blk, 0, stream>>>(xl, xr, ei, ea, att, alpha, amax, E);
  k_edge_exp  <<<(E * HH + blk - 1) / blk, blk, 0, stream>>>(ei, alpha, amax, denom, E);
  k_edge_agg  <<<(E + 7) / 8, blk, 0, stream>>>(xl, ei, alpha, denom, out, E);
}
